// MemSystem_68324339745365
// MI455X (gfx1250) — compile-verified, matrix operand order checked by reading, not run on hardware
//
#include <hip/hip_runtime.h>

// ---------------------------------------------------------------------------
// MI455X (gfx1250) implementation. wave32, WMMA 16x16x32 bf16 (f32 accum),
// TDM (tensor_load_to_lds) staging for the final read GEMM.
// B=1024, D=256, R=16384, W=64, K(conv)=3
// ---------------------------------------------------------------------------

#define BB 1024
#define DD 256
#define RR 16384
#define WW 64
#define KSTEP 128   // mem2 rows staged into LDS per TDM transfer (128*128B = 16KB)

typedef __attribute__((ext_vector_type(16))) __bf16 v16bf;
typedef __attribute__((ext_vector_type(8)))  float  v8f;
typedef __attribute__((ext_vector_type(4)))  unsigned int v4u;
typedef __attribute__((ext_vector_type(4)))  int    v4i;
typedef __attribute__((ext_vector_type(8)))  int    v8i;

#ifndef USE_TDM
#define USE_TDM 1
#endif
#if USE_TDM && __has_builtin(__builtin_amdgcn_tensor_load_to_lds)
#define TDM_PATH 1
#else
#define TDM_PATH 0
#endif

// K-index inside a 16-element bf16 A/B fragment (ISA 7.12.2, 16-bit A 16x32):
// lanes 0-15 hold K 0..7 (VGPR0-3) and 16..23 (VGPR4-7); lanes 16-31 hold +8.
__device__ __forceinline__ int frag_k_idx(int e, int half) {
    int v = e >> 1;                       // VGPR index 0..7
    return ((v & 4) ? 16 : 0) + 2 * (v & 3) + (e & 1) + 8 * half;
}

// Fragment: per-lane index selects a row, K contiguous in memory
// (A row-major [M,K], or B given as B^T row-major [N,K]).
__device__ __forceinline__ v16bf frag_m_major(const __bf16* p, long ld, int idx, int half) {
    v16bf f;
    const __bf16* row = p + (long)idx * ld;
#pragma unroll
    for (int e = 0; e < 16; ++e) f[e] = row[frag_k_idx(e, half)];
    return f;
}

// Fragment: K strides through memory (B stored [K,N] row-major, or A stored [K,M]).
__device__ __forceinline__ v16bf frag_k_major(const __bf16* p, long ld, int idx, int half) {
    v16bf f;
#pragma unroll
    for (int e = 0; e < 16; ++e) f[e] = p[(long)frag_k_idx(e, half) * ld + idx];
    return f;
}

__device__ __forceinline__ float softplus_f(float z) {
    return (z > 20.f) ? z : log1pf(__expf(z));
}

// ---------------------------------------------------------------------------
// Controller heads: v = x@Wv+bv (store bf16), vn = ||v||, beta, gamma.
// ---------------------------------------------------------------------------
__global__ void heads_kernel(const float* __restrict__ x,
                             const float* __restrict__ Wv, const float* __restrict__ bv,
                             const float* __restrict__ Wb, const float* __restrict__ bb,
                             const float* __restrict__ Wg, const float* __restrict__ bg,
                             __bf16* __restrict__ v_bf, float* __restrict__ beta,
                             float* __restrict__ gamma, float* __restrict__ vn) {
    int b = blockIdx.x, t = threadIdx.x;              // t = output column 0..63
    const float* xr = x + (long)b * DD;
    float acc = bv[t];
    for (int d = 0; d < DD; ++d) acc += xr[d] * Wv[(long)d * WW + t];
    v_bf[(long)b * WW + t] = (__bf16)acc;

    __shared__ float red[64];
    red[t] = acc * acc; __syncthreads();
    for (int s = 32; s > 0; s >>= 1) { if (t < s) red[t] += red[t + s]; __syncthreads(); }
    if (t == 0) vn[b] = sqrtf(red[0]);
    __syncthreads();

    float pb = 0.f;
    for (int d = t; d < DD; d += 64) pb += xr[d] * Wb[d];
    red[t] = pb; __syncthreads();
    for (int s = 32; s > 0; s >>= 1) { if (t < s) red[t] += red[t + s]; __syncthreads(); }
    if (t == 0) beta[b] = softplus_f(red[0] + bb[0]);
    __syncthreads();

    float pg = 0.f;
    for (int d = t; d < DD; d += 64) pg += xr[d] * Wg[d];
    red[t] = pg; __syncthreads();
    for (int s = 32; s > 0; s >>= 1) { if (t < s) red[t] += red[t + s]; __syncthreads(); }
    if (t == 0) gamma[b] = 1.f + softplus_f(red[0] + bg[0]);
}

// mn[r] = ||mem[r,:]||, plus bf16 copy of mem.
__global__ void memrow_kernel(const float* __restrict__ mem,
                              __bf16* __restrict__ mem_bf, float* __restrict__ mn) {
    int r = blockIdx.x, t = threadIdx.x;
    float m = mem[(long)r * WW + t];
    mem_bf[(long)r * WW + t] = (__bf16)m;
    __shared__ float red[64];
    red[t] = m * m; __syncthreads();
    for (int s = 32; s > 0; s >>= 1) { if (t < s) red[t] += red[t + s]; __syncthreads(); }
    if (t == 0) mn[r] = sqrtf(red[0]);
}

__global__ void cvt_bf16_kernel(const float* __restrict__ in, __bf16* __restrict__ out, long n) {
    long i = (long)blockIdx.x * 256 + threadIdx.x;
    if (i < n) out[i] = (__bf16)in[i];
}

// ---------------------------------------------------------------------------
// GEMM1: logits_w[b,r] = beta[b] * (v . mem_r) / (vn[b]*mn[r] + eps)
// ---------------------------------------------------------------------------
__global__ void sim_logits_kernel(const __bf16* __restrict__ vbf,
                                  const __bf16* __restrict__ membf,
                                  const float* __restrict__ beta,
                                  const float* __restrict__ vn,
                                  const float* __restrict__ mn,
                                  float* __restrict__ logits) {
    const int NT = RR / 16;
    int wave = threadIdx.x >> 5, lane = threadIdx.x & 31;
    int tile = blockIdx.x * 4 + wave;
    int mt = tile / NT, nt = tile % NT;
    int m0 = mt * 16, n0 = nt * 16;
    int half = lane >> 4, li = lane & 15;
    v8f acc = {};
#pragma unroll
    for (int kk = 0; kk < WW; kk += 32) {
        v16bf a  = frag_m_major(vbf + (long)m0 * WW + kk, WW, li, half);
        v16bf bf = frag_m_major(membf + (long)n0 * WW + kk, WW, li, half); // B^T = mem rows
        acc = __builtin_amdgcn_wmma_f32_16x16x32_bf16(false, a, false, bf,
                                                      (short)0, acc, false, false);
    }
    int r = n0 + li;
    float mnr = mn[r];
#pragma unroll
    for (int i = 0; i < 8; ++i) {
        int b = m0 + i + 8 * half;
        // fast reciprocal (v_rcp_f32) instead of IEEE division: softmax argument
        float s = acc[i] * __builtin_amdgcn_rcpf(vn[b] * mnr + 1e-16f);
        logits[(long)b * RR + r] = beta[b] * s;
    }
}

// ---------------------------------------------------------------------------
// Row softmax over R columns. One 256-thread block per row.
// ---------------------------------------------------------------------------
__global__ void softmax_rows_kernel(float* __restrict__ buf, __bf16* __restrict__ out_bf) {
    int row = blockIdx.x, t = threadIdx.x;
    float* p = buf + (long)row * RR;
    __shared__ float red[256];

    float m = -3.4e38f;
    for (int c = t; c < RR; c += 256) m = fmaxf(m, p[c]);
    red[t] = m; __syncthreads();
    for (int s = 128; s > 0; s >>= 1) { if (t < s) red[t] = fmaxf(red[t], red[t + s]); __syncthreads(); }
    float rowmax = red[0]; __syncthreads();

    float sum = 0.f;
    for (int c = t; c < RR; c += 256) sum += __expf(p[c] - rowmax);
    red[t] = sum; __syncthreads();
    for (int s = 128; s > 0; s >>= 1) { if (t < s) red[t] += red[t + s]; __syncthreads(); }
    float inv = 1.f / red[0]; __syncthreads();

    for (int c = t; c < RR; c += 256) {
        float e = __expf(p[c] - rowmax) * inv;
        if (out_bf) out_bf[(long)row * RR + c] = (__bf16)e;
        else        p[c] = e;
    }
}

// ---------------------------------------------------------------------------
// a = conv1d_same(w, k)^gamma, renormalized; write f32 + bf16 copies.
// ---------------------------------------------------------------------------
__global__ void conv_pow_norm_kernel(const float* __restrict__ w,
                                     const float* __restrict__ conv_k,
                                     const float* __restrict__ conv_b,
                                     const float* __restrict__ gamma,
                                     float* __restrict__ a, __bf16* __restrict__ a_bf) {
    int b = blockIdx.x, t = threadIdx.x;
    const float* wr = w + (long)b * RR;
    float* ar = a + (long)b * RR;
    float k0 = conv_k[0], k1 = conv_k[1], k2 = conv_k[2], cb = conv_b[0];
    float g = gamma[b];
    __shared__ float red[256];

    float sum = 0.f;
    for (int r = t; r < RR; r += 256) {
        float wl = (r > 0)      ? wr[r - 1] : 0.f;
        float wm = wr[r];
        float wg = (r < RR - 1) ? wr[r + 1] : 0.f;
        float wc = k0 * wl + k1 * wm + k2 * wg + cb;
        float av = powf(fmaxf(wc, 0.f), g);
        ar[r] = av;
        sum += av;
    }
    red[t] = sum; __syncthreads();
    for (int s = 128; s > 0; s >>= 1) { if (t < s) red[t] += red[t + s]; __syncthreads(); }
    float inv = 1.f / (red[0] + (float)RR * 1e-16f);
    __syncthreads();
    for (int r = t; r < RR; r += 256) {
        float av = ar[r] * inv;
        ar[r] = av;
        a_bf[(long)b * RR + r] = (__bf16)av;
    }
}

// erase[r] = mean_b a[b,r]  (coalesced; prefetch the streaming column walk)
__global__ void erase_kernel(const float* __restrict__ a, float* __restrict__ erase) {
    int r = blockIdx.x * 256 + threadIdx.x;
    if (r >= RR) return;
    float s = 0.f;
    for (int b = 0; b < BB; ++b) {
        if (b + 8 < BB) __builtin_prefetch(&a[(long)(b + 8) * RR + r], 0, 1);
        s += a[(long)b * RR + r];
    }
    erase[r] = s * (1.f / BB);
}

// ---------------------------------------------------------------------------
// GEMM2 fused: mem2 = mem*(1-erase) + (a^T @ v)/B   (bf16 out). M=R,N=W,K=B.
// ---------------------------------------------------------------------------
__global__ void add_mem2_kernel(const __bf16* __restrict__ abf,
                                const __bf16* __restrict__ vbf,
                                const float* __restrict__ mem,
                                const float* __restrict__ erase,
                                __bf16* __restrict__ mem2bf) {
    const int NT = WW / 16;
    int wave = threadIdx.x >> 5, lane = threadIdx.x & 31;
    int tile = blockIdx.x * 4 + wave;
    int mt = tile / NT, nt = tile % NT;
    int m0 = mt * 16, n0 = nt * 16;
    int half = lane >> 4, li = lane & 15;
    v8f acc = {};
    for (int kk = 0; kk < BB; kk += 32) {
        v16bf a = frag_k_major(abf + (long)kk * RR + m0, RR, li, half); // A[m][k]=a[k,m]
        v16bf b = frag_k_major(vbf + (long)kk * WW + n0, WW, li, half); // B[k][n]=v[k,n]
        acc = __builtin_amdgcn_wmma_f32_16x16x32_bf16(false, a, false, b,
                                                      (short)0, acc, false, false);
    }
    int w = n0 + li;
#pragma unroll
    for (int i = 0; i < 8; ++i) {
        int r = m0 + i + 8 * half;
        float val = mem[(long)r * WW + w] * (1.f - erase[r]) + acc[i] * (1.f / BB);
        mem2bf[(long)r * WW + w] = (__bf16)val;
    }
}

// ---------------------------------------------------------------------------
// GEMM3: logits_p = x @ Wp + bp.   M=B, N=R, K=D=256.
// ---------------------------------------------------------------------------
__global__ void p_logits_kernel(const __bf16* __restrict__ xbf,
                                const __bf16* __restrict__ Wpbf,
                                const float* __restrict__ bp,
                                float* __restrict__ logits) {
    const int NT = RR / 16;
    int wave = threadIdx.x >> 5, lane = threadIdx.x & 31;
    int tile = blockIdx.x * 4 + wave;
    int mt = tile / NT, nt = tile % NT;
    int m0 = mt * 16, n0 = nt * 16;
    int half = lane >> 4, li = lane & 15;
    v8f acc = {};
#pragma unroll
    for (int kk = 0; kk < DD; kk += 32) {
        v16bf a = frag_m_major(xbf + (long)m0 * DD + kk, DD, li, half);
        v16bf b = frag_k_major(Wpbf + (long)kk * RR + n0, RR, li, half);
        acc = __builtin_amdgcn_wmma_f32_16x16x32_bf16(false, a, false, b,
                                                      (short)0, acc, false, false);
    }
    int r = n0 + li;
    float bias = bp[r];
#pragma unroll
    for (int i = 0; i < 8; ++i) {
        int b = m0 + i + 8 * half;
        logits[(long)b * RR + r] = acc[i] + bias;
    }
}

// ---------------------------------------------------------------------------
// GEMM4: out = p @ mem2.  M=B, N=W=64, K=R=16384.
// One block per 16-row band of p; 4 waves cover the 4 N-tiles.
// mem2 K-slabs staged into LDS by the Tensor Data Mover (double buffered),
// WMMA B-fragments then read from LDS.
// ---------------------------------------------------------------------------
#if TDM_PATH
// Build a 2-D D# (ISA 08_async_tensor §8) and issue TENSOR_LOAD_TO_LDS:
// tensor = [rows_remaining x 64] bf16 at gsrc, tile = [KSTEP x 64], dst = LDS.
// This toolchain exposes the 6-arg builtin:
//   (uint32x4 g0, int32x8 g1, int32x4 g2, int32x4 g3, int32x8 gx, i32 cpol)
__device__ __forceinline__ void tdm_load_slab(const __bf16* gsrc, unsigned lds_off,
                                              int tile_rows, unsigned rows_remaining) {
    unsigned long long ga = (unsigned long long)(uintptr_t)gsrc;
    v4u g0;
    g0[0] = 1u;                                            // count=1, user descriptor
    g0[1] = lds_off;                                       // lds_addr (bytes)
    g0[2] = (unsigned)(ga & 0xFFFFFFFFu);                  // global_addr[31:0]
    g0[3] = (unsigned)((ga >> 32) & 0x01FFFFFFu) | (2u << 30); // addr[56:32] | type=2
    v8i g1;
    g1[0] = 0x00010000;                                    // data_size=1 (2 bytes)
    g1[1] = (int)((unsigned)WW << 16);                     // tensor_dim0=64 (bits 63:48)
    g1[2] = (int)((rows_remaining & 0xFFFFu) << 16);       // tensor_dim1 lo16 (bit 80)
    g1[3] = (int)(rows_remaining >> 16) | (int)((unsigned)WW << 16); // dim1 hi16 | tile_dim0
    g1[4] = tile_rows;                                     // tile_dim1 (bits 143:128)
    g1[5] = WW;                                            // tensor_dim0_stride lo32
    g1[6] = 0;
    g1[7] = 0;
    v4i z4 = {0, 0, 0, 0};                                 // 2-D: groups 2/3 unused
    v8i z8 = {0, 0, 0, 0, 0, 0, 0, 0};
    __builtin_amdgcn_tensor_load_to_lds(g0, g1, z4, z4, z8, 0);
}
__device__ __forceinline__ unsigned lds_offset_of(const void* p) {
    // generic LDS pointer: low 32 bits are the wave-relative LDS byte offset
    return (unsigned)(uintptr_t)p;
}
#endif

__global__ void out_kernel(const __bf16* __restrict__ pbf,
                           const __bf16* __restrict__ mem2bf,
                           float* __restrict__ out) {
    __shared__ __align__(16) __bf16 bt[2][KSTEP][WW];      // 2 x 16KB
    const int NS = RR / KSTEP;                             // 128 slabs
    int tid = threadIdx.x;
    int wave = tid >> 5, lane = tid & 31;
    int m0 = blockIdx.x * 16;
    int n0 = wave * 16;
    int half = lane >> 4, li = lane & 15;
    v8f acc = {};

#if TDM_PATH
    if (tid < 32) tdm_load_slab(mem2bf, lds_offset_of(&bt[0][0][0]), KSTEP, RR);
    for (int s = 0; s < NS; ++s) {
        int buf = s & 1;
        if (tid < 32) {
            if (s + 1 < NS) {
                tdm_load_slab(mem2bf + (long)(s + 1) * KSTEP * WW,
                              lds_offset_of(&bt[(s + 1) & 1][0][0]),
                              KSTEP, (unsigned)(RR - (s + 1) * KSTEP));
                __builtin_amdgcn_s_wait_tensorcnt(1);      // slab s landed
            } else {
                __builtin_amdgcn_s_wait_tensorcnt(0);
            }
        }
        __syncthreads();                                   // LDS visible to all waves
#pragma unroll
        for (int kk = 0; kk < KSTEP; kk += 32) {
            v16bf a = frag_m_major(pbf + (long)m0 * RR + s * KSTEP + kk, RR, li, half);
            v16bf b;
#pragma unroll
            for (int e = 0; e < 16; ++e) b[e] = bt[buf][kk + frag_k_idx(e, half)][n0 + li];
            acc = __builtin_amdgcn_wmma_f32_16x16x32_bf16(false, a, false, b,
                                                          (short)0, acc, false, false);
        }
        __syncthreads();                                   // done reading before reuse
    }
#else
    for (int s = 0; s < NS; ++s) {
        // synchronous cooperative staging fallback
        const uint4* gs = (const uint4*)(mem2bf + (long)s * KSTEP * WW);
        uint4* ls = (uint4*)&bt[0][0][0];
        for (int i = tid; i < KSTEP * WW * 2 / 16; i += 128) ls[i] = gs[i];
        __syncthreads();
#pragma unroll
        for (int kk = 0; kk < KSTEP; kk += 32) {
            v16bf a = frag_m_major(pbf + (long)m0 * RR + s * KSTEP + kk, RR, li, half);
            v16bf b;
#pragma unroll
            for (int e = 0; e < 16; ++e) b[e] = bt[0][kk + frag_k_idx(e, half)][n0 + li];
            acc = __builtin_amdgcn_wmma_f32_16x16x32_bf16(false, a, false, b,
                                                          (short)0, acc, false, false);
        }
        __syncthreads();
    }
#endif

    int w = n0 + li;
#pragma unroll
    for (int i = 0; i < 8; ++i) {
        int b = m0 + i + 8 * half;
        out[(long)b * WW + w] = acc[i];
    }
}

// ---------------------------------------------------------------------------
// Host driver
// ---------------------------------------------------------------------------
extern "C" void kernel_launch(void* const* d_in, const int* in_sizes, int n_in,
                              void* d_out, int out_size, void* d_ws, size_t ws_size,
                              hipStream_t stream) {
    (void)in_sizes; (void)n_in; (void)out_size; (void)ws_size;
    const float* x      = (const float*)d_in[0];
    const float* Wv     = (const float*)d_in[1];
    const float* bv     = (const float*)d_in[2];
    const float* Wb     = (const float*)d_in[3];
    const float* bb     = (const float*)d_in[4];
    const float* Wg     = (const float*)d_in[5];
    const float* bg     = (const float*)d_in[6];
    const float* Wp     = (const float*)d_in[7];
    const float* bp     = (const float*)d_in[8];
    const float* conv_k = (const float*)d_in[9];
    const float* conv_b = (const float*)d_in[10];
    const float* mem    = (const float*)d_in[11];

    char* ws = (char*)d_ws;
    size_t off = 0;
    auto take = [&](size_t bytes) -> char* {
        char* p = ws + off;
        off += (bytes + 255) & ~(size_t)255;
        return p;
    };

    __bf16* v_bf    = (__bf16*)take((size_t)BB * WW * 2);
    __bf16* x_bf    = (__bf16*)take((size_t)BB * DD * 2);
    __bf16* mem_bf  = (__bf16*)take((size_t)RR * WW * 2);
    __bf16* mem2_bf = (__bf16*)take((size_t)RR * WW * 2);
    __bf16* Wp_bf   = (__bf16*)take((size_t)DD * RR * 2);
    float*  beta    = (float*)take((size_t)BB * 4);
    float*  gamma   = (float*)take((size_t)BB * 4);
    float*  vn      = (float*)take((size_t)BB * 4);
    float*  mn      = (float*)take((size_t)RR * 4);
    float*  erase   = (float*)take((size_t)RR * 4);
    float*  logits  = (float*)take((size_t)BB * RR * 4);  // reused: w logits -> w -> p logits
    float*  a_f     = (float*)take((size_t)BB * RR * 4);
    __bf16* a_bf    = (__bf16*)take((size_t)BB * RR * 2);
    __bf16* p_bf    = (__bf16*)take((size_t)BB * RR * 2);

    // 1. heads + norms + casts
    heads_kernel<<<BB, 64, 0, stream>>>(x, Wv, bv, Wb, bb, Wg, bg, v_bf, beta, gamma, vn);
    memrow_kernel<<<RR, 64, 0, stream>>>(mem, mem_bf, mn);
    cvt_bf16_kernel<<<((long)BB * DD + 255) / 256, 256, 0, stream>>>(x, x_bf, (long)BB * DD);
    cvt_bf16_kernel<<<((long)DD * RR + 255) / 256, 256, 0, stream>>>(Wp, Wp_bf, (long)DD * RR);

    // 2. content addressing: WMMA sim + softmax (in place)
    sim_logits_kernel<<<(BB / 16) * (RR / 16) / 4, 128, 0, stream>>>(v_bf, mem_bf, beta, vn, mn, logits);
    softmax_rows_kernel<<<BB, 256, 0, stream>>>(logits, nullptr);

    // 3. shift-conv + sharpen + renormalize
    conv_pow_norm_kernel<<<BB, 256, 0, stream>>>(logits, conv_k, conv_b, gamma, a_f, a_bf);
    erase_kernel<<<(RR + 255) / 256, 256, 0, stream>>>(a_f, erase);

    // 4. memory write (WMMA a^T@v, fused erase)
    add_mem2_kernel<<<(RR / 16) * (WW / 16) / 4, 128, 0, stream>>>(a_bf, v_bf, mem, erase, mem2_bf);

    // 5. read address: WMMA x@Wp + softmax -> bf16 p
    p_logits_kernel<<<(BB / 16) * (RR / 16) / 4, 128, 0, stream>>>(x_bf, Wp_bf, bp, logits);
    softmax_rows_kernel<<<BB, 256, 0, stream>>>(logits, p_bf);

    // 6. read: WMMA p@mem2 with TDM-staged B operand
    out_kernel<<<BB / 16, 128, 0, stream>>>(p_bf, mem2_bf, (float*)d_out);
}
